// HierarchicalCSNet_45268955300462
// MI455X (gfx1250) — matrix operand order repaired and checked by reading, not compiled, not run on hardware
//
#include <hip/hip_runtime.h>

typedef __attribute__((ext_vector_type(16))) _Float16 v16h;
typedef __attribute__((ext_vector_type(8)))  _Float16 v8h;
typedef __attribute__((ext_vector_type(8)))  float    v8f;
typedef __attribute__((ext_vector_type(4)))  int      v4i;

#define HW   256
#define NPIX 65536

#if __has_builtin(__builtin_amdgcn_global_load_async_to_lds_b128)
#define HAVE_ASYNC_LDS 1
#endif

__device__ __forceinline__ void wait_async0() {
#if defined(HAVE_ASYNC_LDS)
#if __has_builtin(__builtin_amdgcn_s_wait_asynccnt)
  __builtin_amdgcn_s_wait_asynccnt(0);
#else
  asm volatile("s_wait_asynccnt 0x0" ::: "memory");
#endif
#endif
}

__device__ __forceinline__ v16h cat16(v8h lo, v8h hi) {
  v16h r;
#pragma unroll
  for (int i = 0; i < 8; ++i) { r[i] = lo[i]; r[i + 8] = hi[i]; }
  return r;
}

__device__ __forceinline__ int iclamp(int v, int lo, int hi) {
  return v < lo ? lo : (v > hi ? hi : v);
}

// ---------------------------------------------------------------------------
// Weight repack: [CoutTotal][Cin][3][3] f32  ->  [CoutTotal][9][Cin] f16
// ---------------------------------------------------------------------------
__global__ void pack_w(const float* __restrict__ W, _Float16* __restrict__ Wp,
                       int Cin, int CoutTotal) {
  int idx = blockIdx.x * blockDim.x + threadIdx.x;
  int per = Cin * 9;
  int total = CoutTotal * per;
  if (idx >= total) return;
  int m   = idx / per;
  int rem = idx - m * per;
  int rs  = rem / Cin;
  int ci  = rem - rs * Cin;
  Wp[idx] = (_Float16)W[(size_t)m * per + (size_t)ci * 9 + rs];
}

// ---------------------------------------------------------------------------
// Sampling: conv(1->13, k=32, s=32) on 256x256 -> s[13][8][8]
// ---------------------------------------------------------------------------
__global__ void sampling(const float* __restrict__ x, const float* __restrict__ sw,
                         float* __restrict__ s) {
  int idx = blockIdx.x * blockDim.x + threadIdx.x;
  if (idx >= 13 * 64) return;
  int k  = idx >> 6;
  int ij = idx & 63;
  int i = ij >> 3, j = ij & 7;
  const float* w  = sw + k * 1024;
  const float* xp = x + (i * 32) * HW + j * 32;
  float acc = 0.0f;
  for (int p = 0; p < 32; ++p) {
#pragma unroll 8
    for (int q = 0; q < 32; ++q)
      acc += w[p * 32 + q] * xp[p * HW + q];
  }
  s[idx] = acc;
}

// ---------------------------------------------------------------------------
// 1x1 conv 13->1024 (+bias) fused with pixel-shuffle(32): r[256][256] (f32)
// ---------------------------------------------------------------------------
__global__ void up_reshape(const float* __restrict__ s, const float* __restrict__ uw,
                           const float* __restrict__ ub, float* __restrict__ rout) {
  int idx = blockIdx.x * blockDim.x + threadIdx.x;  // 65536
  int y = idx >> 8, xx = idx & 255;
  int i = y >> 5, p = y & 31, j = xx >> 5, q = xx & 31;
  int c = p * 32 + q;
  float acc = ub[c];
#pragma unroll
  for (int k = 0; k < 13; ++k)
    acc += uw[c * 13 + k] * s[k * 64 + i * 8 + j];
  rout[idx] = acc;
}

// ---------------------------------------------------------------------------
// Head conv: 1->64, 3x3 pad 1, + bias + PReLU -> f16 (branch-free padding)
// ---------------------------------------------------------------------------
__global__ void conv1to64(const float* __restrict__ rin, const float* __restrict__ w,
                          const float* __restrict__ b, const float* __restrict__ a,
                          _Float16* __restrict__ out) {
  int idx = blockIdx.x * blockDim.x + threadIdx.x;  // 64*65536
  int m   = idx >> 16;
  int pix = idx & 65535;
  int y = pix >> 8, xx = pix & 255;
  float acc = b[m];
#pragma unroll
  for (int r = 0; r < 3; ++r) {
#pragma unroll
    for (int s = 0; s < 3; ++s) {
      int yy = y + r - 1, xc = xx + s - 1;
      float v = rin[iclamp(yy, 0, HW - 1) * HW + iclamp(xc, 0, HW - 1)];
      bool ok = (yy >= 0) && (yy < HW) && (xc >= 0) && (xc < HW);
      acc += w[m * 9 + r * 3 + s] * (ok ? v : 0.0f);
    }
  }
  float al = *a;
  acc = (acc >= 0.0f) ? acc : al * acc;
  out[idx] = (_Float16)acc;
}

// ---------------------------------------------------------------------------
// Tail conv: 64->1, 3x3 pad 1, + bias; f16 in -> f32 out (branch-free)
// ---------------------------------------------------------------------------
__global__ void conv64to1(const _Float16* __restrict__ in, const float* __restrict__ w,
                          const float* __restrict__ b, float* __restrict__ out) {
  int idx = blockIdx.x * blockDim.x + threadIdx.x;  // 65536
  int y = idx >> 8, xx = idx & 255;
  float acc = *b;
  for (int c = 0; c < 64; ++c) {
#pragma unroll
    for (int r = 0; r < 3; ++r) {
#pragma unroll
      for (int s = 0; s < 3; ++s) {
        int yy = y + r - 1, xc = xx + s - 1;
        float v = (float)in[(size_t)c * NPIX + iclamp(yy, 0, HW - 1) * HW +
                            iclamp(xc, 0, HW - 1)];
        bool ok = (yy >= 0) && (yy < HW) && (xc >= 0) && (xc < HW);
        acc += w[c * 9 + r * 3 + s] * (ok ? v : 0.0f);
      }
    }
  }
  out[idx] = acc;
}

// ---------------------------------------------------------------------------
// Heavy conv: CIN(64|128) -> 64, 3x3 pad 1, + bias + PReLU, f16 activations.
// Implicit GEMM with V_WMMA_F32_16X16X32_F16.
// Per (row-tap rr, 32-ch slab): 3 A-chunks async-to-LDS + one shared
// 66-pixel halo window [n][k]; core pixels fetched with 2x b128 per thread
// (uniform rowOK branch), only 2 halo pixels use clamped u16+select.
// Compute: per tap, 4 B fragments batch-loaded then 4 WMMAs (overlapped
// dscnt waits). 12 WMMAs per barrier pair.
// ---------------------------------------------------------------------------
template <int CIN>
__global__ void __launch_bounds__(128)
conv3x3_wmma(const _Float16* __restrict__ inA, const _Float16* __restrict__ inB,
             const _Float16* __restrict__ Wp, const float* __restrict__ bias,
             const float* __restrict__ alpha, _Float16* __restrict__ out) {
  constexpr int K    = 9 * CIN;
  constexpr int CPRS = CIN / 32;   // 32-ch slabs per tap

  __shared__ __align__(16) _Float16 As[3 * 64 * 32];  // [sx][m][k]
  __shared__ __align__(16) _Float16 Win[68 * 32];     // [window n][k]

  const int tid   = threadIdx.x;
  const int lane  = tid & 31;
  const int wave  = tid >> 5;
  const int tileY = blockIdx.x >> 2;
  const int x0    = (blockIdx.x & 3) << 6;

  v8f acc[4] = {};

  const int half = lane >> 4;
  const int col  = lane & 15;
  const int mrow = (wave << 4) + col;
  const int kk   = tid & 31;
  const int wgrp = tid >> 5;        // 0..3: 16-pixel core segment owner
  const int arow = tid >> 1;        // A-stage: row per thread-pair
  const int aseg = (tid & 1) << 4;  // 16 halves per thread

  for (int rr = 0; rr < 3; ++rr) {
    const int  yy    = tileY + rr - 1;
    const int  yc    = iclamp(yy, 0, HW - 1);
    const bool rowOK = (yy >= 0) && (yy < HW);   // block-uniform

    for (int cc = 0; cc < CPRS; ++cc) {
      const int c0 = cc << 5;

      // --- stage A: 3 taps of this row, 64x32 f16 each (async to LDS) ---
#pragma unroll
      for (int sx = 0; sx < 3; ++sx) {
        const _Float16* src =
            Wp + (size_t)arow * K + (rr * 3 + sx) * CIN + c0 + aseg;
        _Float16* dst = &As[sx * 2048 + arow * 32 + aseg];
#if defined(HAVE_ASYNC_LDS)
        __builtin_amdgcn_global_load_async_to_lds_b128(
            (__attribute__((address_space(1))) v4i*)(void*)src,
            (__attribute__((address_space(3))) v4i*)dst, 0, 0);
        __builtin_amdgcn_global_load_async_to_lds_b128(
            (__attribute__((address_space(1))) v4i*)(void*)(src + 8),
            (__attribute__((address_space(3))) v4i*)(dst + 8), 0, 0);
#else
        *(uint4*)dst       = *(const uint4*)src;
        *(uint4*)(dst + 8) = *(const uint4*)(src + 8);
#endif
      }

      // --- stage 66-pixel halo window, transposed [n][k] ---
      {
        const int c = c0 + kk;
        const _Float16* src =
            (CIN == 64) ? (inA + (size_t)c * NPIX)
                        : ((c < 64) ? (inA + (size_t)c * NPIX)
                                    : (inB + (size_t)(c - 64) * NPIX));
        const _Float16* srcRow = src + (size_t)yc * HW;

        // 16 aligned core pixels per thread (x in [x0+wgrp*16, +16) always
        // in-bounds in x); uniform rowOK branch -> scalar s_cbranch only.
        union { uint4 q[2]; _Float16 h[16]; } d;
        d.q[0] = make_uint4(0u, 0u, 0u, 0u);
        d.q[1] = make_uint4(0u, 0u, 0u, 0u);
        if (rowOK) {
          const uint4* p = (const uint4*)(srcRow + x0 + (wgrp << 4));
          d.q[0] = p[0];
          d.q[1] = p[1];
        }
        const int nb = (wgrp << 4) + 1;     // window rows 1..64 = core
#pragma unroll
        for (int i = 0; i < 16; ++i)
          Win[(nb + i) * 32 + kk] = d.h[i];

        // halo pixels: wave 0 -> left (n=0), wave 1 -> right (n=65)
        if (wgrp < 2) {
          const int n  = (wgrp == 0) ? 0 : 65;
          const int xg = x0 - 1 + n;
          _Float16 v = srcRow[iclamp(xg, 0, HW - 1)];
          bool ok = rowOK && (xg >= 0) && (xg < HW);
          Win[n * 32 + kk] = ok ? v : (_Float16)0.0f;
        }
      }
      wait_async0();
      __syncthreads();

      // --- 3 taps x 4 N-subtiles = 12 WMMAs per barrier pair ---
#pragma unroll
      for (int sx = 0; sx < 3; ++sx) {
        const _Float16* ap = &As[sx * 2048 + mrow * 32];
        v8h alo = *(const v8h*)(ap + half * 8);
        v8h ahi = *(const v8h*)(ap + 16 + half * 8);
        v16h a = cat16(alo, ahi);

        v16h bfr[4];
#pragma unroll
        for (int ns = 0; ns < 4; ++ns) {
          int nwin = (ns << 4) + col + sx;   // window row = pixel + tap shift
          const _Float16* bp = &Win[nwin * 32 + half * 8];
          bfr[ns] = cat16(*(const v8h*)bp, *(const v8h*)(bp + 16));
        }
#pragma unroll
        for (int ns = 0; ns < 4; ++ns)
          acc[ns] = __builtin_amdgcn_wmma_f32_16x16x32_f16(
              false, a, false, bfr[ns], (short)0, acc[ns], false, false);
      }
      __syncthreads();
    }
  }

  // --- epilogue: bias + PReLU; D layout: M = half*8+v, N = col ---
  const float al = *alpha;
#pragma unroll
  for (int ns = 0; ns < 4; ++ns) {
    int xcol = x0 + (ns << 4) + col;
#pragma unroll
    for (int v = 0; v < 8; ++v) {
      int m = (wave << 4) + (half << 3) + v;
      float val = acc[ns][v] + bias[m];
      val = (val >= 0.0f) ? val : al * val;
      out[(size_t)m * NPIX + (size_t)tileY * HW + xcol] = (_Float16)val;
    }
  }
}

// ---------------------------------------------------------------------------
// Host orchestration
// ---------------------------------------------------------------------------
extern "C" void kernel_launch(void* const* d_in, const int* in_sizes, int n_in,
                              void* d_out, int out_size, void* d_ws, size_t ws_size,
                              hipStream_t stream) {
  const float* x        = (const float*)d_in[0];
  const float* sample_w = (const float*)d_in[1];
  const float* up_w     = (const float*)d_in[2];
  const float* up_b     = (const float*)d_in[3];
  const float* h1_w     = (const float*)d_in[4];
  const float* h1_b     = (const float*)d_in[5];
  const float* h1_a     = (const float*)d_in[6];
  const float* fus_w    = (const float*)d_in[7];
  const float* fus_b    = (const float*)d_in[8];
  const float* fus_a    = (const float*)d_in[9];
  const float* t2_w     = (const float*)d_in[10];
  const float* t2_b     = (const float*)d_in[11];
  const float* t2_a     = (const float*)d_in[12];
  const float* t3_w     = (const float*)d_in[13];
  const float* t3_b     = (const float*)d_in[14];
  const float* t3_a     = (const float*)d_in[15];
  const float* t4_w     = (const float*)d_in[16];
  const float* t4_b     = (const float*)d_in[17];
  const float* t4_a     = (const float*)d_in[18];
  const float* t5_w     = (const float*)d_in[19];
  const float* t5_b     = (const float*)d_in[20];
  float* out = (float*)d_out;

  char* ws = (char*)d_ws;
  size_t off = 0;
  auto alloc = [&](size_t bytes) -> void* {
    void* p = (void*)(ws + off);
    off += (bytes + 255) & ~(size_t)255;
    return p;
  };

  _Float16* WpT2  = (_Float16*)alloc((size_t)8 * 64 * 576  * sizeof(_Float16));
  _Float16* WpT3  = (_Float16*)alloc((size_t)8 * 64 * 576  * sizeof(_Float16));
  _Float16* WpT4  = (_Float16*)alloc((size_t)8 * 64 * 576  * sizeof(_Float16));
  _Float16* WpFus = (_Float16*)alloc((size_t)7 * 64 * 1152 * sizeof(_Float16));
  float*    sbuf  = (float*)alloc((size_t)13 * 64 * sizeof(float));
  float*    rbuf  = (float*)alloc((size_t)NPIX * sizeof(float));
  _Float16* hbuf  = (_Float16*)alloc((size_t)64 * NPIX * sizeof(_Float16)); // also tail ping
  _Float16* featA = (_Float16*)alloc((size_t)64 * NPIX * sizeof(_Float16));
  _Float16* featB = (_Float16*)alloc((size_t)64 * NPIX * sizeof(_Float16));
  _Float16* tbuf  = (_Float16*)alloc((size_t)64 * NPIX * sizeof(_Float16));

  // One-time (per call) f16 weight repack
  pack_w<<<dim3(1152), dim3(256), 0, stream>>>(t2_w, WpT2, 64, 8 * 64);
  pack_w<<<dim3(1152), dim3(256), 0, stream>>>(t3_w, WpT3, 64, 8 * 64);
  pack_w<<<dim3(1152), dim3(256), 0, stream>>>(t4_w, WpT4, 64, 8 * 64);
  pack_w<<<dim3(2016), dim3(256), 0, stream>>>(fus_w, WpFus, 128, 7 * 64);

  _Float16* featCur = featA;
  _Float16* featNxt = featB;
  for (int m = 0; m < 8; ++m) {
    sampling<<<dim3(4), dim3(256), 0, stream>>>(
        x, sample_w + (size_t)m * 13 * 1024, sbuf);
    up_reshape<<<dim3(256), dim3(256), 0, stream>>>(
        sbuf, up_w + (size_t)m * 1024 * 13, up_b + (size_t)m * 1024, rbuf);

    _Float16* hDst = (m == 0) ? featCur : hbuf;
    conv1to64<<<dim3(16384), dim3(256), 0, stream>>>(
        rbuf, h1_w + (size_t)m * 64 * 9, h1_b + (size_t)m * 64, h1_a + m, hDst);

    if (m > 0) {
      conv3x3_wmma<128><<<dim3(1024), dim3(128), 0, stream>>>(
          featCur, hbuf, WpFus + (size_t)(m - 1) * 64 * 1152,
          fus_b + (size_t)(m - 1) * 64, fus_a + (m - 1), featNxt);
      _Float16* tmp = featCur; featCur = featNxt; featNxt = tmp;
    }

    conv3x3_wmma<64><<<dim3(1024), dim3(128), 0, stream>>>(
        featCur, featCur, WpT2 + (size_t)m * 64 * 576,
        t2_b + (size_t)m * 64, t2_a + m, tbuf);
    conv3x3_wmma<64><<<dim3(1024), dim3(128), 0, stream>>>(
        tbuf, tbuf, WpT3 + (size_t)m * 64 * 576,
        t3_b + (size_t)m * 64, t3_a + m, hbuf);
    conv3x3_wmma<64><<<dim3(1024), dim3(128), 0, stream>>>(
        hbuf, hbuf, WpT4 + (size_t)m * 64 * 576,
        t4_b + (size_t)m * 64, t4_a + m, tbuf);

    conv64to1<<<dim3(256), dim3(256), 0, stream>>>(
        tbuf, t5_w + (size_t)m * 64 * 9, t5_b + m, out + (size_t)m * NPIX);
  }

  (void)in_sizes; (void)n_in; (void)out_size; (void)ws_size;
}
